// MultiHeadAttention_47029891891703
// MI455X (gfx1250) — matrix operand order repaired and checked by reading, not compiled, not run on hardware
//
#include <hip/hip_runtime.h>
#include <hip/hip_bf16.h>

// ---------------- problem constants ----------------
#define B_    4
#define LSEQ  2048       // LQ == LK
#define DM_   1024
#define DKV   64         // DK == DV
#define H_    16

typedef __attribute__((ext_vector_type(16))) _Float16 v16h;
typedef __attribute__((ext_vector_type(8)))  _Float16 v8h;
typedef __attribute__((ext_vector_type(8)))  float    v8f;
typedef __attribute__((ext_vector_type(4)))  float    v4f;
typedef __attribute__((ext_vector_type(4)))  unsigned int v4u;
typedef __attribute__((ext_vector_type(8)))  int      v8i;
typedef __attribute__((ext_vector_type(4)))  int      v4i;

static __device__ __forceinline__ v8f wmma16x16x32(v16h a, v16h b, v8f c) {
    // D = A(16x32) * B(32x16) + C   (f16 in, f32 accum)
    return __builtin_amdgcn_wmma_f32_16x16x32_f16(
        /*neg_a=*/false, a, /*neg_b=*/false, b,
        /*c_mod=*/(short)0, c, /*reuse_a=*/false, /*reuse_b=*/false);
}

// ---------------- TDM (Tensor Data Mover) support ----------------
#define HAVE_TDM __has_builtin(__builtin_amdgcn_tensor_load_to_lds)

template <int N>
static __device__ __forceinline__ void wait_tensorcnt() {
#if __has_builtin(__builtin_amdgcn_s_wait_tensorcnt)
    __builtin_amdgcn_s_wait_tensorcnt((short)N);
#else
    asm volatile("s_wait_tensorcnt %0" :: "i"(N) : "memory");
#endif
}

#if HAVE_TDM
// 2D f16 tile load: global (td0 x td1 tensor, row stride stride0 elements)
// -> contiguous LDS tile (tile1 rows of tile0 elements).  D# per ISA 8.3/8.4.
static __device__ __forceinline__ void tdm_load_2d_f16(
    unsigned lds_addr, const void* gptr,
    unsigned td0, unsigned td1, unsigned tile0, unsigned tile1,
    unsigned long long stride0 /* elements */)
{
    unsigned long long ga = (unsigned long long)(uintptr_t)gptr;
    v4u g0;
    g0[0] = 1u;                                           // count=1, user mode
    g0[1] = lds_addr;                                     // LDS byte address
    g0[2] = (unsigned)(ga & 0xFFFFFFFFull);               // global_addr lo
    g0[3] = (unsigned)((ga >> 32) & 0x01FFFFFFull) | (2u << 30); // hi + type=2
    v8i g1;
    g1[0] = (int)(1u << 16);                              // data_size = 2 bytes
    g1[1] = (int)((td0 & 0xFFFFu) << 16);                 // tensor_dim0[15:0]
    g1[2] = (int)(((td0 >> 16) & 0xFFFFu) | ((td1 & 0xFFFFu) << 16));
    g1[3] = (int)(((td1 >> 16) & 0xFFFFu) | ((tile0 & 0xFFFFu) << 16));
    g1[4] = (int)(tile1 & 0xFFFFu);                       // tile_dim1, tile_dim2=0
    g1[5] = (int)(unsigned)(stride0 & 0xFFFFFFFFull);     // dim0 stride lo
    g1[6] = (int)(unsigned)((stride0 >> 32) & 0xFFFFull); // stride hi, dim1 stride=0
    g1[7] = 0;
    v4i z4 = {};
#if defined(__clang_major__) && (__clang_major__ >= 23)
    v8i z8 = {};
    __builtin_amdgcn_tensor_load_to_lds(g0, g1, z4, z4, z8, 0);
#else
    __builtin_amdgcn_tensor_load_to_lds(g0, g1, z4, z4, 0);
#endif
}
#endif

// ---------------------------------------------------------------------------
// Kernel 0: fp32 -> f16 transpose:  dst[h][c][r] = (f16) src[h][r][c]
// ---------------------------------------------------------------------------
__global__ __launch_bounds__(256) void cvt_transpose(
    const float* __restrict__ src, _Float16* __restrict__ dst,
    int Hn, int R, int C)
{
    size_t total = (size_t)Hn * R * C;
    size_t stride = (size_t)gridDim.x * blockDim.x;
    for (size_t idx = (size_t)blockIdx.x * blockDim.x + threadIdx.x;
         idx < total; idx += stride) {
        size_t hh  = idx / ((size_t)R * C);
        size_t rem = idx - hh * (size_t)R * C;
        size_t r = rem / C, c = rem - r * C;
        dst[hh * (size_t)R * C + c * (size_t)R + r] = (_Float16)src[idx];
    }
}

// ---------------------------------------------------------------------------
// Kernel 1: projections. z=0/1/2 -> {Q,K,V}[b,h] = src[b] x W^T[h] (f16 WMMA).
// Qh/Kh row-major (L x 64) f16; V stored transposed Vt (64 x L) f16.
// ---------------------------------------------------------------------------
__global__ __launch_bounds__(256) void proj_kernel(
    const float* __restrict__ qin, const float* __restrict__ kin,
    const float* __restrict__ vin,
    const _Float16* __restrict__ wqT, const _Float16* __restrict__ wkT,
    const _Float16* __restrict__ wvT,
    _Float16* __restrict__ Qh, _Float16* __restrict__ Kh,
    _Float16* __restrict__ Vt)
{
    const int z    = blockIdx.z;            // 0:Q 1:K 2:V
    const int bh   = blockIdx.y;            // b*H + h
    const int b    = bh >> 4;
    const int h    = bh & 15;
    const int m0   = blockIdx.x * 64;
    const int lane = threadIdx.x & 31;
    const int w    = threadIdx.x >> 5;
    const int lrow = lane & 15, lhi = lane >> 4;
    const int mi   = w & 3;
    const int ni0  = (w >> 2) * 2;

    const float*    src = (z == 0) ? qin : (z == 1) ? kin : vin;
    const _Float16* W   = (z == 0) ? wqT : (z == 1) ? wkT : wvT;

    const int row = m0 + mi * 16 + lrow;
    const float*    srcRow = src + ((size_t)b * LSEQ + row) * DM_;
    const _Float16* Wh     = W + (size_t)h * DKV * DM_;

    v8f c0 = {}, c1 = {};
    for (int kk = 0; kk < DM_; kk += 32) {
        const int kb = kk + lhi * 16;
        v4f f0 = *(const v4f*)(srcRow + kb);
        v4f f1 = *(const v4f*)(srcRow + kb + 4);
        v4f f2 = *(const v4f*)(srcRow + kb + 8);
        v4f f3 = *(const v4f*)(srcRow + kb + 12);
        union { v16h v; v8h p[2]; } a;
        v8h lo, hi;
        #pragma unroll
        for (int i = 0; i < 4; ++i) {
            lo[i]     = (_Float16)f0[i];
            lo[4 + i] = (_Float16)f1[i];
            hi[i]     = (_Float16)f2[i];
            hi[4 + i] = (_Float16)f3[i];
        }
        a.p[0] = lo; a.p[1] = hi;
        const int kbb = kk + lhi * 16;
        v16h b0 = *(const v16h*)(Wh + (size_t)(ni0 * 16 + lrow) * DM_ + kbb);
        v16h b1 = *(const v16h*)(Wh + (size_t)(ni0 * 16 + 16 + lrow) * DM_ + kbb);
        c0 = wmma16x16x32(a.v, b0, c0);
        c1 = wmma16x16x32(a.v, b1, c1);
    }

    if (z < 2) {
        _Float16* dst = ((z == 0) ? Qh : Kh) + (size_t)bh * LSEQ * DKV;
        #pragma unroll
        for (int r = 0; r < 8; ++r) {
            const int M = m0 + mi * 16 + lhi * 8 + r;
            dst[(size_t)M * DKV + ni0 * 16 + lrow]      = (_Float16)c0[r];
            dst[(size_t)M * DKV + ni0 * 16 + 16 + lrow] = (_Float16)c1[r];
        }
    } else {
        _Float16* dst = Vt + (size_t)bh * DKV * LSEQ;
        v8h o0, o1;
        #pragma unroll
        for (int r = 0; r < 8; ++r) { o0[r] = (_Float16)c0[r]; o1[r] = (_Float16)c1[r]; }
        const int Mb = m0 + mi * 16 + lhi * 8;
        *(v8h*)(dst + (size_t)(ni0 * 16 + lrow) * LSEQ + Mb)      = o0;
        *(v8h*)(dst + (size_t)(ni0 * 16 + 16 + lrow) * LSEQ + Mb) = o1;
    }
}

// ---------------------------------------------------------------------------
// Kernel 2: fused flash-style attention per (b,h, 64 q-rows).
//  - TDM double-buffers the 64x64 K-tile and V^T-tile into LDS (TENSORcnt)
//  - WMMA S = scale * Q K^T; raw S -> HBM attn region
//  - online softmax (LDS row reductions), WMMA acc += P * V^T  (ds_load frags)
//  - final in-kernel sweep normalizes attn from L2-resident raw scores
// ---------------------------------------------------------------------------
__global__ __launch_bounds__(256) void attn_kernel(
    const _Float16* __restrict__ Qh, const _Float16* __restrict__ Kh,
    const _Float16* __restrict__ Vt, _Float16* __restrict__ ctx,
    float* __restrict__ attn)
{
    __shared__ float    Sv[64][72];
    __shared__ _Float16 Pb[64][80];
    __shared__ _Float16 KtBuf[2][64 * 64];   // K rows kt..kt+63, row-major (64)
    __shared__ _Float16 VtBuf[2][64 * 64];   // V^T rows dv=0..63, cols kt..kt+63
    __shared__ float    m_run[64], l_run[64], factor[64], pred[64][4];

    const int bh   = blockIdx.y;
    const int q0   = blockIdx.x * 64;
    const int lane = threadIdx.x & 31;
    const int w    = threadIdx.x >> 5;
    const int lrow = lane & 15, lhi = lane >> 4;
    const int mi   = w & 3;
    const int ni0  = (w >> 2) * 2;

    const _Float16* Qb = Qh + (size_t)bh * LSEQ * DKV;
    const _Float16* Kb = Kh + (size_t)bh * LSEQ * DKV;
    const _Float16* Vb = Vt + (size_t)bh * DKV * LSEQ;
    float* attnB = attn + (size_t)bh * LSEQ * LSEQ + (size_t)q0 * LSEQ;

    // Preload the wave's Q fragments (rows fixed for whole kernel, K=64 -> 2 steps)
    union { v16h v; v8h p[2]; } aq[2];
    const _Float16* qrow = Qb + (size_t)(q0 + mi * 16 + lrow) * DKV;
    #pragma unroll
    for (int ks = 0; ks < 2; ++ks) {
        aq[ks].p[0] = *(const v8h*)(qrow + ks * 32 + lhi * 16);
        aq[ks].p[1] = *(const v8h*)(qrow + ks * 32 + lhi * 16 + 8);
    }

    if (threadIdx.x < 64) { m_run[threadIdx.x] = -1e30f; l_run[threadIdx.x] = 0.0f; }
    v8f acc0 = {}, acc1 = {};

#if HAVE_TDM
    // issue DMA for tile 0 (wave 0 only; EXEC-independent wave-level op)
    if (w == 0) {
        tdm_load_2d_f16((unsigned)(uintptr_t)&KtBuf[0][0], Kb,
                        DKV, LSEQ, 64, 64, DKV);
        tdm_load_2d_f16((unsigned)(uintptr_t)&VtBuf[0][0], Vb,
                        LSEQ, DKV, 64, 64, LSEQ);
    }
#endif
    __syncthreads();

    const float scale = 0.125f;  // 1/sqrt(64)

    for (int kt = 0; kt < LSEQ; kt += 64) {
        const int buf = (kt >> 6) & 1;
#if HAVE_TDM
        if (w == 0) {
            if (kt + 64 < LSEQ) {
                // prefetch next tile into the other buffer, then wait for current
                tdm_load_2d_f16((unsigned)(uintptr_t)&KtBuf[buf ^ 1][0],
                                Kb + (size_t)(kt + 64) * DKV,
                                DKV, LSEQ, 64, 64, DKV);
                tdm_load_2d_f16((unsigned)(uintptr_t)&VtBuf[buf ^ 1][0],
                                Vb + (size_t)(kt + 64),
                                LSEQ, DKV, 64, 64, LSEQ);
                wait_tensorcnt<2>();
            } else {
                wait_tensorcnt<0>();
            }
        }
#else
        // fallback: cooperative copy of the current tiles into LDS
        for (int idx = threadIdx.x; idx < 512; idx += 256)
            ((v8h*)&KtBuf[buf][0])[idx] =
                *(const v8h*)(Kb + (size_t)kt * DKV + (size_t)idx * 8);
        for (int idx = threadIdx.x; idx < 512; idx += 256) {
            const int rr = idx >> 3, cc = (idx & 7) * 8;
            *(v8h*)&VtBuf[buf][rr * 64 + cc] =
                *(const v8h*)(Vb + (size_t)rr * LSEQ + kt + cc);
        }
#endif
        __syncthreads();   // LDS tiles (and m/l init on first iter) visible

        // ---- S = scale * Q K^T  (B fragments from LDS K tile) ----
        const _Float16* Ktile = &KtBuf[buf][0];
        v8f s0 = {}, s1 = {};
        #pragma unroll
        for (int ks = 0; ks < 2; ++ks) {
            v16h b0 = *(const v16h*)(Ktile + (ni0 * 16 + lrow) * 64 + ks * 32 + lhi * 16);
            v16h b1 = *(const v16h*)(Ktile + (ni0 * 16 + 16 + lrow) * 64 + ks * 32 + lhi * 16);
            s0 = wmma16x16x32(aq[ks].v, b0, s0);
            s1 = wmma16x16x32(aq[ks].v, b1, s1);
        }
        // stage to LDS + write raw scaled scores to HBM (attn scratch)
        #pragma unroll
        for (int r = 0; r < 8; ++r) {
            const int M = mi * 16 + lhi * 8 + r;
            const float v0 = s0[r] * scale, v1 = s1[r] * scale;
            Sv[M][ni0 * 16 + lrow]      = v0;
            Sv[M][ni0 * 16 + 16 + lrow] = v1;
            attnB[(size_t)M * LSEQ + kt + ni0 * 16 + lrow]      = v0;
            attnB[(size_t)M * LSEQ + kt + ni0 * 16 + 16 + lrow] = v1;
        }
        __syncthreads();

        // ---- row max (4 threads / row) ----
        {
            const int rr = threadIdx.x >> 2, seg = threadIdx.x & 3;
            float mx = -1e30f;
            #pragma unroll
            for (int j = 0; j < 16; ++j) mx = fmaxf(mx, Sv[rr][seg * 16 + j]);
            pred[rr][seg] = mx;
        }
        __syncthreads();
        if (threadIdx.x < 64) {
            const int rr = threadIdx.x;
            float mx = fmaxf(fmaxf(pred[rr][0], pred[rr][1]),
                             fmaxf(pred[rr][2], pred[rr][3]));
            float mnew = fmaxf(m_run[rr], mx);
            factor[rr] = __expf(m_run[rr] - mnew);
            m_run[rr]  = mnew;
        }
        __syncthreads();

        // ---- rescale running accumulators ----
        #pragma unroll
        for (int r = 0; r < 8; ++r) {
            const float f = factor[mi * 16 + lhi * 8 + r];
            acc0[r] *= f; acc1[r] *= f;
        }

        // ---- P = exp(S - m) into LDS (f16) + row-sum partials ----
        {
            const int rr = threadIdx.x >> 2, seg = threadIdx.x & 3;
            const float m = m_run[rr];
            float sum = 0.0f;
            #pragma unroll
            for (int j = 0; j < 16; ++j) {
                float e = __expf(Sv[rr][seg * 16 + j] - m);
                Pb[rr][seg * 16 + j] = (_Float16)e;
                sum += e;
            }
            pred[rr][seg] = sum;
        }
        __syncthreads();
        if (threadIdx.x < 64) {
            const int rr = threadIdx.x;
            l_run[rr] = l_run[rr] * factor[rr] +
                        pred[rr][0] + pred[rr][1] + pred[rr][2] + pred[rr][3];
        }
        __syncthreads();

        // ---- acc += P * V^T  (A from LDS P, B from LDS V tile) ----
        const _Float16* Vtile = &VtBuf[buf][0];
        #pragma unroll
        for (int ks = 0; ks < 2; ++ks) {
            union { v16h v; v8h p[2]; } ap;
            const _Float16* prow = &Pb[mi * 16 + lrow][0];
            ap.p[0] = *(const v8h*)(prow + ks * 32 + lhi * 16);
            ap.p[1] = *(const v8h*)(prow + ks * 32 + lhi * 16 + 8);
            v16h b0 = *(const v16h*)(Vtile + (ni0 * 16 + lrow) * 64 + ks * 32 + lhi * 16);
            v16h b1 = *(const v16h*)(Vtile + (ni0 * 16 + 16 + lrow) * 64 + ks * 32 + lhi * 16);
            acc0 = wmma16x16x32(ap.v, b0, acc0);
            acc1 = wmma16x16x32(ap.v, b1, acc1);
        }
        __syncthreads();
    }

    // ---- ctx = acc / l ; layout (B, LQ, H, DV) so reshape matches reference ----
    {
        const int b = bh >> 4, h = bh & 15;
        _Float16* ctxB = ctx + ((size_t)b * LSEQ + q0) * (H_ * DKV) + h * DKV;
        #pragma unroll
        for (int r = 0; r < 8; ++r) {
            const int M = mi * 16 + lhi * 8 + r;
            const float il = 1.0f / l_run[M];
            ctxB[(size_t)M * (H_ * DKV) + ni0 * 16 + lrow]      = (_Float16)(acc0[r] * il);
            ctxB[(size_t)M * (H_ * DKV) + ni0 * 16 + 16 + lrow] = (_Float16)(acc1[r] * il);
        }
    }

    // ---- normalize attn from raw scores (L2-resident: 512 KB per WG) ----
    __threadfence();
    __syncthreads();
    if (threadIdx.x < 64) factor[threadIdx.x] = 1.0f / l_run[threadIdx.x];
    __syncthreads();
    for (int idx = threadIdx.x; idx < 64 * LSEQ; idx += 256) {
        const int rr = idx >> 11;            // LSEQ == 2048
        const int col = idx & (LSEQ - 1);
        const float sraw = attnB[(size_t)rr * LSEQ + col];
        attnB[(size_t)rr * LSEQ + col] = __expf(sraw - m_run[rr]) * factor[rr];
    }
}

// ---------------------------------------------------------------------------
// Kernel 3: out = ctx (8192 x 1024, f16) x wo (1024 x 1024) -> fp32
// ---------------------------------------------------------------------------
__global__ __launch_bounds__(256) void outproj_kernel(
    const _Float16* __restrict__ ctx, const _Float16* __restrict__ woT,
    float* __restrict__ out)
{
    const int m0   = blockIdx.x * 64;        // token rows over B*LQ
    const int n0   = blockIdx.y * 64;        // output dim
    const int lane = threadIdx.x & 31;
    const int w    = threadIdx.x >> 5;
    const int lrow = lane & 15, lhi = lane >> 4;
    const int mi   = w & 3;
    const int ni0  = (w >> 2) * 2;

    const _Float16* arow = ctx + (size_t)(m0 + mi * 16 + lrow) * (H_ * DKV);

    v8f c0 = {}, c1 = {};
    for (int kk = 0; kk < H_ * DKV; kk += 32) {
        union { v16h v; v8h p[2]; } a;
        a.p[0] = *(const v8h*)(arow + kk + lhi * 16);
        a.p[1] = *(const v8h*)(arow + kk + lhi * 16 + 8);
        const int kbb = kk + lhi * 16;
        v16h b0 = *(const v16h*)(woT + (size_t)(n0 + ni0 * 16 + lrow) * (H_ * DKV) + kbb);
        v16h b1 = *(const v16h*)(woT + (size_t)(n0 + ni0 * 16 + 16 + lrow) * (H_ * DKV) + kbb);
        c0 = wmma16x16x32(a.v, b0, c0);
        c1 = wmma16x16x32(a.v, b1, c1);
    }
    #pragma unroll
    for (int r = 0; r < 8; ++r) {
        const int M = m0 + mi * 16 + lhi * 8 + r;
        out[(size_t)M * DM_ + n0 + ni0 * 16 + lrow]      = c0[r];
        out[(size_t)M * DM_ + n0 + ni0 * 16 + 16 + lrow] = c1[r];
    }
}

// ---------------------------------------------------------------------------
// kernel_launch
// inputs (setup_inputs order): q, k, v, wq, wk, wv, wo  (all fp32)
// outputs: out (B*LQ*DM fp32) then attn (B*H*LQ*LK fp32), concatenated
// workspace needed: ~75.5 MB
// ---------------------------------------------------------------------------
extern "C" void kernel_launch(void* const* d_in, const int* in_sizes, int n_in,
                              void* d_out, int out_size, void* d_ws, size_t ws_size,
                              hipStream_t stream) {
    (void)in_sizes; (void)n_in; (void)out_size; (void)ws_size;

    const float* q  = (const float*)d_in[0];
    const float* k  = (const float*)d_in[1];
    const float* v  = (const float*)d_in[2];
    const float* wq = (const float*)d_in[3];
    const float* wk = (const float*)d_in[4];
    const float* wv = (const float*)d_in[5];
    const float* wo = (const float*)d_in[6];

    float* out  = (float*)d_out;
    float* attn = out + (size_t)B_ * LSEQ * DM_;   // out first, then attn

    // workspace carve (f16 elements)
    _Float16* ws  = (_Float16*)d_ws;
    _Float16* wqT = ws;                                   // H*DK*DM   = 1,048,576
    _Float16* wkT = wqT + (size_t)H_ * DKV * DM_;
    _Float16* wvT = wkT + (size_t)H_ * DKV * DM_;
    _Float16* woT = wvT + (size_t)H_ * DKV * DM_;         // DM*(H*DV) = 1,048,576
    _Float16* Qh  = woT + (size_t)DM_ * (H_ * DKV);       // B*H*L*DK  = 8,388,608
    _Float16* Kh  = Qh  + (size_t)B_ * H_ * LSEQ * DKV;
    _Float16* Vt  = Kh  + (size_t)B_ * H_ * LSEQ * DKV;
    _Float16* ctx = Vt  + (size_t)B_ * H_ * LSEQ * DKV;   // B*L*H*DV  = 8,388,608

    // 0) weight convert + transpose to f16
    cvt_transpose<<<dim3(1024), 256, 0, stream>>>(wq, wqT, H_, DM_, DKV);
    cvt_transpose<<<dim3(1024), 256, 0, stream>>>(wk, wkT, H_, DM_, DKV);
    cvt_transpose<<<dim3(1024), 256, 0, stream>>>(wv, wvT, H_, DM_, DKV);
    cvt_transpose<<<dim3(1024), 256, 0, stream>>>(wo, woT, 1, DM_, H_ * DKV);

    // 1) Q/K/V projections (z selects which)
    proj_kernel<<<dim3(LSEQ / 64, B_ * H_, 3), 256, 0, stream>>>(
        q, k, v, wqT, wkT, wvT, Qh, Kh, Vt);

    // 2) fused attention (TDM-staged K/V tiles) + attn normalization
    attn_kernel<<<dim3(LSEQ / 64, B_ * H_), 256, 0, stream>>>(
        Qh, Kh, Vt, ctx, attn);

    // 3) output projection
    outproj_kernel<<<dim3((B_ * LSEQ) / 64, DM_ / 64), 256, 0, stream>>>(
        ctx, woT, out);
}
